// VisionAttention_26061861552532
// MI455X (gfx1250) — compile-verified, matrix operand order checked by reading, not run on hardware
//
#include <hip/hip_runtime.h>
#include <hip/hip_bf16.h>

#define SEQ       8192
#define DIM       1280
#define NUM_HEADS 16
#define HEAD_DIM  80
#define NSEG      8
#define SEG_LEN   (SEQ / NSEG)
#define QKV_N     (3 * DIM)

static constexpr float SCALING = 0.11180339887498948f; // 80^-0.5

typedef _Float16 h4   __attribute__((ext_vector_type(4)));
typedef _Float16 h8   __attribute__((ext_vector_type(8)));
typedef _Float16 v16h __attribute__((ext_vector_type(16)));
typedef float    v8f  __attribute__((ext_vector_type(8)));

// pointer types matching the builtin's signature:
//   param0: 'vector_size(16) int __device__ *'  (addrspace(1) = global)
//   param1: LDS destination (addrspace(3))
typedef int v4i_vs __attribute__((vector_size(16)));
typedef __attribute__((address_space(1))) v4i_vs glb_v4i_t;
typedef __attribute__((address_space(3))) v4i_vs lds_v4i_t;

#if __has_builtin(__builtin_amdgcn_global_load_async_to_lds_b128)
#define HAVE_ASYNC_LDS 1
// async copy: global (f16 data, 16B) -> LDS, tracked by ASYNCcnt
#define ASYNC_CP16(dst_lds, src_glb)                                           \
    __builtin_amdgcn_global_load_async_to_lds_b128(                            \
        (glb_v4i_t*)(unsigned long long)(src_glb),                             \
        (lds_v4i_t*)(unsigned long long)(dst_lds), 0, 0)
#define ASYNC_WAIT() asm volatile("s_wait_asynccnt 0x0" ::: "memory")
#else
#define HAVE_ASYNC_LDS 0
#define ASYNC_CP16(dst_lds, src_glb) (*(h8*)(dst_lds) = *(const h8*)(src_glb))
#define ASYNC_WAIT()
#endif

static __device__ __forceinline__ v16h combine8(h8 lo, h8 hi) {
    return __builtin_shufflevector(lo, hi, 0,1,2,3,4,5,6,7,8,9,10,11,12,13,14,15);
}
static __device__ __forceinline__ v8f wmma_f16(v16h a, v16h b, v8f c) {
    // emits v_wmma_f32_16x16x32_f16
    return __builtin_amdgcn_wmma_f32_16x16x32_f16(false, a, false, b, (short)0, c, false, false);
}

// ---- within-16-lane butterfly reductions (VALU v_permlane16_b32, no LDS) ----
#if __has_builtin(__builtin_amdgcn_permlane16)
static __device__ __forceinline__ float permx(float x, unsigned s0, unsigned s1) {
    return __uint_as_float(__builtin_amdgcn_permlane16(
        __float_as_uint(x), __float_as_uint(x), s0, s1, false, false));
}
static __device__ __forceinline__ float hmax16(float x) {
    x = fmaxf(x, permx(x, 0x67452301u, 0xEFCDAB89u));   // xor 1
    x = fmaxf(x, permx(x, 0x54761032u, 0xDCFE98BAu));   // xor 2
    x = fmaxf(x, permx(x, 0x32107654u, 0xBA98FEDCu));   // xor 4
    x = fmaxf(x, permx(x, 0xFEDCBA98u, 0x76543210u));   // xor 8
    return x;
}
static __device__ __forceinline__ float hsum16(float x) {
    x += permx(x, 0x67452301u, 0xEFCDAB89u);
    x += permx(x, 0x54761032u, 0xDCFE98BAu);
    x += permx(x, 0x32107654u, 0xBA98FEDCu);
    x += permx(x, 0xFEDCBA98u, 0x76543210u);
    return x;
}
#else
static __device__ __forceinline__ float hmax16(float x) {
    #pragma unroll
    for (int d = 1; d < 16; d <<= 1) x = fmaxf(x, __shfl_xor(x, d, 16));
    return x;
}
static __device__ __forceinline__ float hsum16(float x) {
    #pragma unroll
    for (int d = 1; d < 16; d <<= 1) x += __shfl_xor(x, d, 16);
    return x;
}
#endif

// ---------------------------------------------------------------------------
// NT GEMM: C[M,N] = A[M,K] * B[N,K]^T + bias.  A is f32 or f16, B is f32
// (converted to f16 in the tile loader), C is f16 or f32. fp32 accumulate.
// Block: 256 threads (8 waves), tile 128x64, BK=64.
// ---------------------------------------------------------------------------
template<bool A_HALF, bool OUT_HALF>
__global__ __launch_bounds__(256) void gemm_nt_wmma(
    const void* __restrict__ Av, const float* __restrict__ B,
    const float* __restrict__ bias, void* __restrict__ Cv,
    int M, int N, int K)
{
    constexpr int BM = 128, BN = 64, BK = 64, LDT = 72; // 144B rows: 16B aligned, 36-bank step
    __shared__ _Float16 As[BM][LDT];
    __shared__ _Float16 Bs[BN][LDT];

    const int tid  = threadIdx.x;
    const int lane = tid & 31;
    const int w    = tid >> 5;
    const int wm   = w & 3;        // 4 M-groups of 32 rows
    const int wn   = w >> 2;       // 2 N-groups of 32 cols
    const int r    = lane & 15;
    const int kb   = (lane & 16) ? 8 : 0;
    const int m0   = blockIdx.y * BM;
    const int n0   = blockIdx.x * BN;

    const float*    Af = (const float*)Av;
    const _Float16* Ah = (const _Float16*)Av;

    v8f acc[2][2] = {};

    for (int k0 = 0; k0 < K; k0 += BK) {
        // ---- stage A tile (128 x 64) into LDS as f16 ----
        if constexpr (A_HALF) {
            // already f16: byte-for-byte copies -> async to LDS on CDNA5
            #pragma unroll
            for (int p = 0; p < 4; ++p) {
                int row = (tid >> 3) + p * 32;
                int c8  = (tid & 7) * 8;
                ASYNC_CP16(&As[row][c8], Ah + (size_t)(m0 + row) * K + k0 + c8);
            }
        } else {
            #pragma unroll
            for (int p = 0; p < 8; ++p) {
                int row = (tid >> 4) + p * 16;
                int c4  = (tid & 15) * 4;
                float4 v = *(const float4*)(Af + (size_t)(m0 + row) * K + k0 + c4);
                h4 hv = { (_Float16)v.x, (_Float16)v.y, (_Float16)v.z, (_Float16)v.w };
                *(h4*)&As[row][c4] = hv;
            }
        }
        // ---- stage B tile (64 x 64) into LDS as f16 (f32 -> f16 in VGPRs) ----
        #pragma unroll
        for (int p = 0; p < 4; ++p) {
            int row = (tid >> 4) + p * 16;
            int c4  = (tid & 15) * 4;
            float4 v = *(const float4*)(B + (size_t)(n0 + row) * K + k0 + c4);
            h4 hv = { (_Float16)v.x, (_Float16)v.y, (_Float16)v.z, (_Float16)v.w };
            *(h4*)&Bs[row][c4] = hv;
        }
        // CDNA5 prefetch of next K tile (global_prefetch_b8)
        if (k0 + BK < K) {
            __builtin_prefetch(B + (size_t)(n0 + (tid >> 4)) * K + k0 + BK, 0, 1);
            if constexpr (!A_HALF)
                __builtin_prefetch(Af + (size_t)(m0 + (tid >> 4)) * K + k0 + BK, 0, 1);
        }
        if constexpr (A_HALF) { ASYNC_WAIT(); }
        __syncthreads();

        // ---- 2 WMMA K-steps of 32 per LDS refill ----
        #pragma unroll
        for (int ks = 0; ks < BK; ks += 32) {
            v16h af[2], bf[2];
            #pragma unroll
            for (int mi = 0; mi < 2; ++mi) {
                int arow = wm * 32 + mi * 16 + r;
                af[mi] = combine8(*(const h8*)&As[arow][ks + kb],
                                  *(const h8*)&As[arow][ks + kb + 16]);
            }
            #pragma unroll
            for (int ni = 0; ni < 2; ++ni) {
                int brow = wn * 32 + ni * 16 + r;
                bf[ni] = combine8(*(const h8*)&Bs[brow][ks + kb],
                                  *(const h8*)&Bs[brow][ks + kb + 16]);
            }
            #pragma unroll
            for (int mi = 0; mi < 2; ++mi)
                #pragma unroll
                for (int ni = 0; ni < 2; ++ni)
                    acc[mi][ni] = wmma_f16(af[mi], bf[ni], acc[mi][ni]);
        }
        __syncthreads();
    }

    // ---- epilogue: bias add + store (C layout: VGPR i -> row i + 8*(lane>=16), col = lane&15) ----
    const int roff = (lane & 16) ? 8 : 0;
    #pragma unroll
    for (int mi = 0; mi < 2; ++mi) {
        #pragma unroll
        for (int ni = 0; ni < 2; ++ni) {
            int col = n0 + wn * 32 + ni * 16 + r;
            float bv = bias[col];
            #pragma unroll
            for (int i = 0; i < 8; ++i) {
                int row = m0 + wm * 32 + mi * 16 + i + roff;
                float val = acc[mi][ni][i] + bv;
                if constexpr (OUT_HALF)
                    ((_Float16*)Cv)[(size_t)row * N + col] = (_Float16)val;
                else
                    ((float*)Cv)[(size_t)row * N + col] = val;
            }
        }
    }
}

// ---------------------------------------------------------------------------
// RoPE applied in place to the Q and K sections of the f16 QKV buffer.
// One thread per (s, head, j<40); handles the (j, j+40) rotation pair.
// ---------------------------------------------------------------------------
__global__ __launch_bounds__(256) void rope_kernel(
    _Float16* __restrict__ qkv, const float* __restrict__ cosb, const float* __restrict__ sinb)
{
    int idx = blockIdx.x * 256 + threadIdx.x;   // [0, SEQ*16*40)
    int s = idx / (NUM_HEADS * 40);
    int t = idx % (NUM_HEADS * 40);
    int h = t / 40;
    int j = t % 40;
    size_t base = (size_t)s * QKV_N + h * HEAD_DIM + j;
    float c0 = cosb[s * HEAD_DIM + j];
    float s0 = sinb[s * HEAD_DIM + j];
    float c1 = cosb[s * HEAD_DIM + j + 40];
    float s1 = sinb[s * HEAD_DIM + j + 40];
    // Q
    float q0 = (float)qkv[base], q1 = (float)qkv[base + 40];
    qkv[base]      = (_Float16)(q0 * c0 - q1 * s0);
    qkv[base + 40] = (_Float16)(q1 * c1 + q0 * s1);
    // K
    float k0 = (float)qkv[base + DIM], k1 = (float)qkv[base + DIM + 40];
    qkv[base + DIM]      = (_Float16)(k0 * c0 - k1 * s0);
    qkv[base + DIM + 40] = (_Float16)(k1 * c1 + k0 * s1);
}

// ---------------------------------------------------------------------------
// Flash attention per (segment, head). Block = 256 threads = 8 waves; each
// block owns 128 query rows, each wave owns 16. head_dim 80 zero-padded to 96
// so QK^T is 3 WMMA K-steps; keys processed 32 at a time so P(16x32) feeds PV
// WMMA with a full K=32. Online softmax in fp32 registers; row reductions via
// v_permlane16_b32 (VALU). Q/K tiles streamed with async-to-LDS.
// ---------------------------------------------------------------------------
__global__ __launch_bounds__(256) void attn_kernel(
    const _Float16* __restrict__ qkv, _Float16* __restrict__ Ah)
{
    constexpr int LDQK = 104;  // 208B rows: 16B aligned, 52-bank step (conflict-free over 16 lanes)
    constexpr int LDVP = 56;   // 112B rows: 16B aligned, 28-bank step
    __shared__ _Float16 Qs[128][LDQK];
    __shared__ _Float16 Ks[32][LDQK];
    __shared__ _Float16 Vt[HEAD_DIM][LDVP];     // V transposed: Vt[dim][key]
    __shared__ _Float16 Ps[8][16][LDVP];        // per-wave P tile (16 x 32)

    const int tid  = threadIdx.x;
    const int lane = tid & 31;
    const int w    = tid >> 5;
    const int r    = lane & 15;
    const int kb   = (lane & 16) ? 8 : 0;
    const int roff = (lane & 16) ? 8 : 0;
    const int qc   = blockIdx.x;       // 128-row chunk within segment
    const int head = blockIdx.y;
    const int seg  = blockIdx.z;
    const int srow0 = seg * SEG_LEN + qc * 128;

    // ---- load Q tile (128 x 80, async) and zero pad dims [80,96) ----
    for (int c = tid; c < 128 * 10; c += 256) {
        int row = c / 10, dg = c % 10;
        ASYNC_CP16(&Qs[row][dg * 8],
                   qkv + (size_t)(srow0 + row) * QKV_N + head * HEAD_DIM + dg * 8);
    }
    {
        h8 z = {};
        *(h8*)&Qs[tid >> 1][80 + (tid & 1) * 8] = z;
    }
    ASYNC_WAIT();
    __syncthreads();

    // ---- per-wave Q fragments (A-layout, resident for whole key loop) ----
    v16h qf[3];
    #pragma unroll
    for (int t = 0; t < 3; ++t) {
        int arow = w * 16 + r;
        qf[t] = combine8(*(const h8*)&Qs[arow][t * 32 + kb],
                         *(const h8*)&Qs[arow][t * 32 + kb + 16]);
    }

    v8f Oacc[5] = {};
    float mrow[8], lrow[8];
    #pragma unroll
    for (int i = 0; i < 8; ++i) { mrow[i] = -__builtin_inff(); lrow[i] = 0.0f; }

    for (int kb0 = 0; kb0 < SEG_LEN; kb0 += 32) {
        // ---- cooperative load: K chunk (32x80, async) and V chunk transposed ----
        for (int c = tid; c < 32 * 10; c += 256) {
            int key = c / 10, dg = c % 10;
            size_t srow = (size_t)(seg * SEG_LEN + kb0 + key);
            ASYNC_CP16(&Ks[key][dg * 8],
                       qkv + srow * QKV_N + DIM + head * HEAD_DIM + dg * 8);
            h8 vv = *(const h8*)(qkv + srow * QKV_N + 2 * DIM + head * HEAD_DIM + dg * 8);
            #pragma unroll
            for (int i = 0; i < 8; ++i) Vt[dg * 8 + i][key] = vv[i];
        }
        if (tid < 64) {
            h8 z = {};
            *(h8*)&Ks[tid >> 1][80 + (tid & 1) * 8] = z;
        }
        ASYNC_WAIT();
        __syncthreads();

        // ---- preload all 6 K-fragments, then run the WMMA chain ----
        v16h kf0[3], kf1[3];
        #pragma unroll
        for (int t = 0; t < 3; ++t) {
            kf0[t] = combine8(*(const h8*)&Ks[r][t * 32 + kb],
                              *(const h8*)&Ks[r][t * 32 + kb + 16]);
            kf1[t] = combine8(*(const h8*)&Ks[16 + r][t * 32 + kb],
                              *(const h8*)&Ks[16 + r][t * 32 + kb + 16]);
        }
        v8f s0 = {}, s1 = {};
        #pragma unroll
        for (int t = 0; t < 3; ++t) {
            s0 = wmma_f16(qf[t], kf0[t], s0);
            s1 = wmma_f16(qf[t], kf1[t], s1);
        }

        // ---- online softmax; P written to LDS in C-layout ----
        #pragma unroll
        for (int i = 0; i < 8; ++i) {
            float a0 = s0[i] * SCALING;
            float a1 = s1[i] * SCALING;
            float mnew  = fmaxf(mrow[i], hmax16(fmaxf(a0, a1)));
            float alpha = __expf(mrow[i] - mnew);
            float p0 = __expf(a0 - mnew);
            float p1 = __expf(a1 - mnew);
            lrow[i] = lrow[i] * alpha + hsum16(p0 + p1);
            mrow[i] = mnew;
            #pragma unroll
            for (int t = 0; t < 5; ++t) Oacc[t][i] *= alpha;
            Ps[w][i + roff][r]      = (_Float16)p0;
            Ps[w][i + roff][16 + r] = (_Float16)p1;
        }
        // same-wave DS RAW: make the cross-lane P stores visible before A-frag reads
        asm volatile("s_wait_dscnt 0" ::: "memory");

        // ---- O += P @ V : preload P fragment + all 5 V fragments, then WMMAs ----
        v16h pf = combine8(*(const h8*)&Ps[w][r][kb],
                           *(const h8*)&Ps[w][r][kb + 16]);
        v16h bv[5];
        #pragma unroll
        for (int t = 0; t < 5; ++t)
            bv[t] = combine8(*(const h8*)&Vt[t * 16 + r][kb],
                             *(const h8*)&Vt[t * 16 + r][kb + 16]);
        #pragma unroll
        for (int t = 0; t < 5; ++t)
            Oacc[t] = wmma_f16(pf, bv[t], Oacc[t]);
        __syncthreads();
    }

    // ---- finalize: divide by row sums, store f16 [s][head*80+d] ----
    #pragma unroll
    for (int i = 0; i < 8; ++i) {
        float inv = 1.0f / lrow[i];
        int row = srow0 + w * 16 + i + roff;
        #pragma unroll
        for (int t = 0; t < 5; ++t)
            Ah[(size_t)row * DIM + head * HEAD_DIM + t * 16 + r] =
                (_Float16)(Oacc[t][i] * inv);
    }
}

// ---------------------------------------------------------------------------
extern "C" void kernel_launch(void* const* d_in, const int* in_sizes, int n_in,
                              void* d_out, int out_size, void* d_ws, size_t ws_size,
                              hipStream_t stream) {
    const float* hidden = (const float*)d_in[0];
    const float* cosb   = (const float*)d_in[1];
    const float* sinb   = (const float*)d_in[2];
    const float* qkv_w  = (const float*)d_in[3];
    const float* qkv_b  = (const float*)d_in[4];
    const float* proj_w = (const float*)d_in[5];
    const float* proj_b = (const float*)d_in[6];
    float* out = (float*)d_out;

    _Float16* qkv_h  = (_Float16*)d_ws;                                   // [SEQ][3840] f16
    _Float16* attn_h = (_Float16*)((char*)d_ws +
                        (size_t)SEQ * QKV_N * sizeof(_Float16));          // [SEQ][1280] f16

    // 1) QKV = hidden @ qkv_w^T + qkv_b  (f32 in -> f16 out)
    gemm_nt_wmma<false, true><<<dim3(QKV_N / 64, SEQ / 128), 256, 0, stream>>>(
        hidden, qkv_w, qkv_b, qkv_h, SEQ, QKV_N, DIM);

    // 2) RoPE on Q,K in place
    rope_kernel<<<(SEQ * NUM_HEADS * 40) / 256, 256, 0, stream>>>(qkv_h, cosb, sinb);

    // 3) segmented flash attention
    attn_kernel<<<dim3(SEG_LEN / 128, NUM_HEADS, NSEG), 256, 0, stream>>>(qkv_h, attn_h);

    // 4) out = attn @ proj_w^T + proj_b  (f16 A, f32 out)
    gemm_nt_wmma<true, false><<<dim3(DIM / 64, SEQ / 128), 256, 0, stream>>>(
        attn_h, proj_w, proj_b, out, SEQ, DIM, DIM);
}
